// DynamicGRU_3882650436467
// MI455X (gfx1250) — matrix operand order repaired and checked by reading, not compiled, not run on hardware
//
#include <hip/hip_runtime.h>

// DynamicGRU / AGRU: T=200, B=512, I=256, H=256. Unused z-gate dropped.
// Phase 1: GI = X @ W_ih(r,n)^T + b_ih  -> WMMA GEMM, stored f16 in d_ws ([T*B][512]).
// Phase 2: persistent per-batch-tile recurrence; W_hh(r,n) register-resident
//          (loop-invariant B operands), h exchanged via LDS, gi/att loads
//          software-pipelined one step ahead.

#define T_STEPS 200
#define BATCH   512
#define IDIM    256
#define HDIM    256

typedef __attribute__((ext_vector_type(16))) _Float16 v16h;
typedef __attribute__((ext_vector_type(8)))  _Float16 v8h_t;
typedef __attribute__((ext_vector_type(8)))  float    v8f;

// Build a v16h operand from two contiguous 8-half (16B) LDS/global chunks.
__device__ __forceinline__ v16h pack16(const _Float16* p0, const _Float16* p1) {
  v8h_t lo = *(const v8h_t*)p0;
  v8h_t hi = *(const v8h_t*)p1;
  v16h r;
#pragma unroll
  for (int i = 0; i < 8; ++i) { r[i] = lo[i]; r[8 + i] = hi[i]; }
  return r;
}

// GI column j in [0,512): j<256 -> r-gate row j; j>=256 -> n-gate row 256+j (=512+(j-256)).
__device__ __forceinline__ int wih_row(int col) {
  return (col < HDIM) ? col : (HDIM + col);
}

// ---------------- Phase 1: input-projection GEMM ----------------
#define K1_MT  128
#define K1_NT  64
#define K1_LD  264   // padded row stride (halves) to spread LDS banks

extern "C" __global__ __launch_bounds__(256)
void agru_gi_gemm(const float* __restrict__ x,
                  const float* __restrict__ w_ih,
                  const float* __restrict__ b_ih,
                  _Float16* __restrict__ gi) {
  extern __shared__ char smem[];
  _Float16* xl = (_Float16*)smem;                                 // 128 x 264 f16
  _Float16* wl = (_Float16*)(smem + (size_t)K1_MT * K1_LD * 2);   //  64 x 264 f16

  const int tid = threadIdx.x;
  const int bn  = blockIdx.x & 7;     // 8 N-blocks of 64 over 512 gate columns
  const int bm  = blockIdx.x >> 3;    // 800 M-blocks of 128 over T*B rows
  const int m0  = bm * K1_MT;
  const int n0  = bn * K1_NT;

  for (int idx = tid; idx < K1_MT * IDIM; idx += 256) {
    int r = idx >> 8, k = idx & 255;
    xl[r * K1_LD + k] = (_Float16)x[(size_t)(m0 + r) * IDIM + k];
  }
  for (int idx = tid; idx < K1_NT * IDIM; idx += 256) {
    int jj = idx >> 8, k = idx & 255;
    wl[jj * K1_LD + k] = (_Float16)w_ih[(size_t)wih_row(n0 + jj) * IDIM + k];
  }
  __syncthreads();

  const int lane = tid & 31;
  const int wave = tid >> 5;          // 8 waves: each owns 16 M rows
  const int ln   = lane & 15;
  const int kh   = lane >> 4;

  const _Float16* arow = xl + (size_t)(wave * 16 + ln) * K1_LD;
  const _Float16* br[4];
#pragma unroll
  for (int nt = 0; nt < 4; ++nt) br[nt] = wl + (size_t)(nt * 16 + ln) * K1_LD;

  v8f acc[4] = {v8f{}, v8f{}, v8f{}, v8f{}};
#pragma unroll
  for (int kb = 0; kb < IDIM; kb += 32) {
    v16h a = pack16(arow + kb + 16 * kh, arow + kb + 16 * kh + 8);
#pragma unroll
    for (int nt = 0; nt < 4; ++nt) {
      v16h b = pack16(br[nt] + kb + 16 * kh, br[nt] + kb + 16 * kh + 8);
      acc[nt] = __builtin_amdgcn_wmma_f32_16x16x32_f16(
          false, a, false, b, (short)0, acc[nt], false, false);
    }
  }

  float bias[4];
#pragma unroll
  for (int nt = 0; nt < 4; ++nt) bias[nt] = b_ih[wih_row(n0 + nt * 16 + ln)];

  const int mrb = 8 * kh;
#pragma unroll
  for (int i = 0; i < 8; ++i) {
    size_t row = (size_t)(m0 + wave * 16 + mrb + i) * (2 * HDIM);
#pragma unroll
    for (int nt = 0; nt < 4; ++nt)
      gi[row + n0 + nt * 16 + ln] = (_Float16)(acc[nt][i] + bias[nt]);
  }
}

// ---------------- Phase 2: persistent recurrence ----------------
#define BT     16    // batch rows per block
#define LDH16  264   // padded f16 row stride (h, WMMA copy)
#define LDH32  260   // padded f32 row stride (h, master)

extern "C" __global__ __launch_bounds__(512)
void agru_recurrent(const _Float16* __restrict__ gi,
                    const float* __restrict__ att,
                    const int*   __restrict__ lengths,
                    const float* __restrict__ h0,
                    const float* __restrict__ w_hh,
                    const float* __restrict__ b_hh,
                    float* __restrict__ out) {
  __shared__ _Float16 HF[BT * LDH16];   // h, f16 copy for WMMA A operands
  __shared__ float    HS[BT * LDH32];   // h, f32 master

  const int tid = threadIdx.x;
  const int b0  = blockIdx.x * BT;

  for (int idx = tid; idx < BT * HDIM; idx += 512) {
    int m = idx >> 8, k = idx & 255;
    float v = h0[(size_t)(b0 + m) * HDIM + k];
    HS[m * LDH32 + k] = v;
    HF[m * LDH16 + k] = (_Float16)v;
  }
  __syncthreads();

  const int lane = tid & 31;
  const int wave = tid >> 5;          // 16 waves: column tile c = wave
  const int ln   = lane & 15;
  const int kh   = lane >> 4;
  const int j    = wave * 16 + ln;    // gate column 0..255 (this lane's N)
  const int mrb  = 8 * kh;            // C-matrix row base for this lane

  const float brj = b_hh[j];
  const float bnj = b_hh[2 * HDIM + j];
  int lenv[8];
#pragma unroll
  for (int i = 0; i < 8; ++i) lenv[i] = lengths[b0 + mrb + i];

  // Loop-invariant B operands: W_hh r/n rows, register-resident for all 200 steps.
  // Lane = N column -> weight row j; 16 contiguous K elements per operand.
  v16h wrv[8], wnv[8];
  {
    const float* wr_g = w_hh + (size_t)j * HDIM;
    const float* wn_g = w_hh + (size_t)(2 * HDIM + j) * HDIM;
#pragma unroll
    for (int kk = 0; kk < 8; ++kk) {
      const int base = kk * 32 + 16 * kh;
      v16h r, n;
#pragma unroll
      for (int e = 0; e < 16; ++e) {
        r[e] = (_Float16)wr_g[base + e];
        n[e] = (_Float16)wn_g[base + e];
      }
      wrv[kk] = r;
      wnv[kk] = n;
    }
  }

  const _Float16* arow = HF + (size_t)ln * LDH16;  // A lane = M row (lane&15)

  // Pipeline priming: gi/att for t=0.
  float gir_c[8], gin_c[8], av_c[8];
#pragma unroll
  for (int i = 0; i < 8; ++i) {
    size_t row = ((size_t)0 * BATCH + (b0 + mrb + i)) * (size_t)(2 * HDIM);
    gir_c[i] = (float)gi[row + j];
    gin_c[i] = (float)gi[row + HDIM + j];
  }
  {
    const float4* ap = (const float4*)(att + b0 + mrb);
    float4 a0 = ap[0], a1 = ap[1];
    av_c[0]=a0.x; av_c[1]=a0.y; av_c[2]=a0.z; av_c[3]=a0.w;
    av_c[4]=a1.x; av_c[5]=a1.y; av_c[6]=a1.z; av_c[7]=a1.w;
  }

#pragma unroll 1
  for (int t = 0; t < T_STEPS; ++t) {
    // Issue next-step gi/att loads early; they complete under this step's WMMAs.
    const int tn = (t + 1 < T_STEPS) ? (t + 1) : t;
    float gir_n[8], gin_n[8], av_n[8];
#pragma unroll
    for (int i = 0; i < 8; ++i) {
      size_t row = ((size_t)tn * BATCH + (b0 + mrb + i)) * (size_t)(2 * HDIM);
      gir_n[i] = (float)gi[row + j];
      gin_n[i] = (float)gi[row + HDIM + j];
    }
    {
      const float4* ap = (const float4*)(att + (size_t)tn * BATCH + b0 + mrb);
      float4 a0 = ap[0], a1 = ap[1];
      av_n[0]=a0.x; av_n[1]=a0.y; av_n[2]=a0.z; av_n[3]=a0.w;
      av_n[4]=a1.x; av_n[5]=a1.y; av_n[6]=a1.z; av_n[7]=a1.w;
    }

    // gh = h @ W_hh(r,n)^T : A streamed from LDS, B from registers.
    v8f cr = {}, cn = {};
#pragma unroll
    for (int kk = 0; kk < 8; ++kk) {
      const int kb = kk * 32;
      v16h a = pack16(arow + kb + 16 * kh, arow + kb + 16 * kh + 8);
      cr = __builtin_amdgcn_wmma_f32_16x16x32_f16(false, a, false, wrv[kk], (short)0, cr, false, false);
      cn = __builtin_amdgcn_wmma_f32_16x16x32_f16(false, a, false, wnv[kk], (short)0, cn, false, false);
    }

    float hnew[8];
#pragma unroll
    for (int i = 0; i < 8; ++i) {
      int   b    = b0 + mrb + i;
      float ghr  = cr[i] + brj;
      float ghn  = cn[i] + bnj;
      float rst  = 1.0f / (1.0f + __expf(-(gir_c[i] + ghr)));
      float nst  = tanhf(gin_c[i] + rst * ghn);
      float hold = HS[(mrb + i) * LDH32 + j];
      float a    = av_c[i];
      float hy   = (1.0f - a) * hold + a * nst;
      bool  live = t < lenv[i];
      hnew[i] = live ? hy : hold;
      out[((size_t)t * BATCH + b) * HDIM + j] = live ? hy : 0.0f;
    }
    __syncthreads();   // all reads of h done
#pragma unroll
    for (int i = 0; i < 8; ++i) {
      HS[(mrb + i) * LDH32 + j] = hnew[i];
      HF[(mrb + i) * LDH16 + j] = (_Float16)hnew[i];
    }
    __syncthreads();   // new h visible

#pragma unroll
    for (int i = 0; i < 8; ++i) {
      gir_c[i] = gir_n[i];
      gin_c[i] = gin_n[i];
      av_c[i]  = av_n[i];
    }
  }
}

extern "C" void kernel_launch(void* const* d_in, const int* in_sizes, int n_in,
                              void* d_out, int out_size, void* d_ws, size_t ws_size,
                              hipStream_t stream) {
  (void)in_sizes; (void)n_in; (void)out_size; (void)ws_size;
  const float* x    = (const float*)d_in[0];
  const float* att  = (const float*)d_in[1];
  const int*   lens = (const int*)  d_in[2];
  const float* h0   = (const float*)d_in[3];
  const float* wih  = (const float*)d_in[4];
  const float* whh  = (const float*)d_in[5];
  const float* bih  = (const float*)d_in[6];
  const float* bhh  = (const float*)d_in[7];
  float*    out = (float*)d_out;
  _Float16* gi  = (_Float16*)d_ws;   // [T*B][512] f16 = 104,857,600 bytes

  const size_t lds1 = (size_t)(K1_MT * K1_LD + K1_NT * K1_LD) * 2;   // ~99 KB
  (void)hipFuncSetAttribute((const void*)agru_gi_gemm,
                            hipFuncAttributeMaxDynamicSharedMemorySize, (int)lds1);

  const int m_blocks = (T_STEPS * BATCH) / K1_MT;   // 800
  agru_gi_gemm<<<dim3(m_blocks * 8), 256, lds1, stream>>>(x, wih, bih, gi);
  agru_recurrent<<<dim3(BATCH / BT), 512, 0, stream>>>(gi, att, lens, h0, whh, bhh, out);
}
